// Block_73615739454081
// MI455X (gfx1250) — matrix operand order repaired and checked by reading, not compiled, not run on hardware
//
#include <hip/hip_runtime.h>
#include <hip/hip_bf16.h>
#include <cstdint>

// ---------------- problem constants ----------------
#define NN   40962      // valid rows
#define DD   512        // d_model
#define HH   4          // heads
#define HDQ  128        // head dim
#define BSZ  256        // block size
#define NBLK 161        // number of blocks
#define NP   41216      // padded rows (NBLK*BSZ)
#define CC   16         // conditioning dim
#define FFW  2048
#define ATT_SCALE 0.08838834764831843f   // 128^-0.5

typedef __attribute__((ext_vector_type(16))) __bf16 bf16x16;
typedef __attribute__((ext_vector_type(8)))  float  f32x8;
typedef __attribute__((ext_vector_type(4)))  unsigned int u32x4;

union Frag16 { bf16x16 v; u32x4 q[2]; };

__device__ __forceinline__ int imin(int a, int b){ return a < b ? a : b; }
__device__ __forceinline__ int imax(int a, int b){ return a > b ? a : b; }

__device__ __forceinline__ f32x8 wmma_bf16(bf16x16 a, bf16x16 b, f32x8 c) {
  // D = A(16x32) * B(32x16) + C, bf16 in / f32 out
  return __builtin_amdgcn_wmma_f32_16x16x32_bf16(
      /*neg_a=*/false, a, /*neg_b=*/false, b,
      /*c_mod=*/(short)0, c, /*reuse_a=*/false, /*reuse_b=*/false);
}

// A fragment: row-major A[lda], lane holds row (lane&15), K-halves per (lane>>4)
__device__ __forceinline__ bf16x16 load_a_frag(const __bf16* A, int lda,
                                               int row0, int k0, int lane) {
  int m = lane & 15, half = lane >> 4;
  const __bf16* p = A + (size_t)(row0 + m) * lda + k0 + half * 8;
  Frag16 f;
  f.q[0] = *(const u32x4*)(p);
  f.q[1] = *(const u32x4*)(p + 16);
  return f.v;
}

// async copy of 16 bytes from global to LDS (ASYNCcnt-tracked, no VGPR data)
__device__ __forceinline__ void async_g2l_b128(const __bf16* gptr, void* lptr) {
  unsigned lds = (unsigned)(uintptr_t)lptr;   // low 32 bits = LDS offset
  asm volatile("global_load_async_to_lds_b128 %0, %1, off"
               :: "v"(lds), "v"(gptr) : "memory");
}
__device__ __forceinline__ void wait_async_le1() {
  asm volatile("s_wait_asynccnt 0x1" ::: "memory");
}
__device__ __forceinline__ void wait_async_0() {
  asm volatile("s_wait_asynccnt 0x0" ::: "memory");
}

__device__ __forceinline__ float gelu_approx(float x) {
  float x3 = x * x * x;
  return 0.5f * x * (1.0f + tanhf(0.7978845608028654f * (x + 0.044715f * x3)));
}

// ---------------- tiny prep kernels ----------------

// so[j] = gnc @ w_cond + b_cond  (j in [0,2D))
__global__ void cond_kernel(const float* __restrict__ gnc,
                            const float* __restrict__ w_cond,
                            const float* __restrict__ b_cond,
                            float* __restrict__ so) {
  int j = blockIdx.x * blockDim.x + threadIdx.x;
  if (j >= 2 * DD) return;
  float s = b_cond[j];
  #pragma unroll
  for (int c = 0; c < CC; ++c) s += gnc[c] * w_cond[c * 2 * DD + j];
  so[j] = s;
}

// out[n][k] = (bf16) in[k][n]   (weight cast + transpose to output-major)
__global__ void transpose_cast(const float* __restrict__ in,
                               __bf16* __restrict__ out, int K, int N) {
  int idx = blockIdx.x * blockDim.x + threadIdx.x;
  if (idx >= K * N) return;
  int k = idx / N, n = idx - k * N;
  out[(size_t)n * K + k] = (__bf16)in[idx];
}

// LayerNorm + conditioning -> bf16, zero-pad rows [Nv, NPr)
__global__ __launch_bounds__(256) void ln_cond(const float* __restrict__ x,
                                               const float* __restrict__ so,
                                               __bf16* __restrict__ h,
                                               int Nv, int NPr) {
  int lane = threadIdx.x & 31, w = threadIdx.x >> 5;
  int row = blockIdx.x * 8 + w;
  if (row >= NPr) return;
  __bf16* hr = h + (size_t)row * DD;
  if (row >= Nv) {
    u32x4 z = {0u, 0u, 0u, 0u};
    *(u32x4*)(hr + lane * 16) = z;
    *(u32x4*)(hr + lane * 16 + 8) = z;
    return;
  }
  const float* xr = x + (size_t)row * DD;
  float vals[16];
  float s = 0.f;
  #pragma unroll
  for (int i = 0; i < 16; ++i) { vals[i] = xr[lane + 32 * i]; s += vals[i]; }
  #pragma unroll
  for (int m = 16; m >= 1; m >>= 1) s += __shfl_xor(s, m, 32);
  float mu = s * (1.0f / DD);
  float var = 0.f;
  #pragma unroll
  for (int i = 0; i < 16; ++i) { float d = vals[i] - mu; var += d * d; }
  #pragma unroll
  for (int m = 16; m >= 1; m >>= 1) var += __shfl_xor(var, m, 32);
  float rs = rsqrtf(var * (1.0f / DD) + 1e-5f);
  #pragma unroll
  for (int i = 0; i < 16; ++i) {
    int c = lane + 32 * i;
    float hn = (vals[i] - mu) * rs;
    hr[c] = (__bf16)(hn * (so[c] + 1.0f) + so[DD + c]);
  }
}

// ---------------- generic WMMA GEMM with async-LDS B staging ----------------
// C[M,Nn] = A[M,K](bf16,row-major) * Bt[Nn,K](bf16,output-major)
// B tile (64 cols x 32 k = 4KB) is double-buffered in LDS via
// global_load_async_to_lds_b128 (ASYNCcnt), shared by all 8 waves.
// Steady-state loop is branch-free; last k-step is peeled.
// modes: 0 bf16 store; 1 bf16 transposed store (vT);
//        2 f32 = resid + acc + bias (row<Mvalid);
//        3 bf16 = gelu(acc+bias); 4 f32 += acc + bias (row<Mvalid)
__global__ __launch_bounds__(256) void gemm_bf16(
    const __bf16* __restrict__ A, const __bf16* __restrict__ Bt,
    __bf16* __restrict__ outb, float* __restrict__ outf,
    const float* __restrict__ resid, const float* __restrict__ bias,
    int M, int Nn, int K, int Mvalid, int mode) {
  __shared__ __align__(16) __bf16 bbuf[2][64 * 32];
  int tid = threadIdx.x;
  int lane = tid & 31, w = tid >> 5;
  int row0 = blockIdx.x * 128 + w * 16;     // M is a multiple of 128: no early exit
  int col0 = blockIdx.y * 64;
  int nrel = lane & 15, kh = lane >> 4;

  // each thread owns one 16B chunk of the B tile: col = tid/4, 8-elem segment
  int colB = tid >> 2, segB = tid & 3;
  const __bf16* gB = Bt + (size_t)(col0 + colB) * K + segB * 8;
  __bf16* lB0 = &bbuf[0][colB * 32 + segB * 8];
  __bf16* lB1 = &bbuf[1][colB * 32 + segB * 8];

  f32x8 acc[4] = {};

  // hoist all 4 B fragments, then run the WMMA chain back-to-back
  auto compute = [&](const __bf16* bb, int k0) {
    bf16x16 a = load_a_frag(A, K, row0, k0, lane);
    Frag16 f[4];
    #pragma unroll
    for (int t = 0; t < 4; ++t) {           // clause of 8 ds_load_b128
      const __bf16* p = bb + (t * 16 + nrel) * 32 + kh * 8;
      f[t].q[0] = *(const u32x4*)(p);
      f[t].q[1] = *(const u32x4*)(p + 16);
    }
    #pragma unroll
    for (int t = 0; t < 4; ++t) acc[t] = wmma_bf16(a, f[t].v, acc[t]);
  };

  async_g2l_b128(gB, lB0);                  // stage tile 0
  const int nt = K / 32;
  for (int i = 0; i < nt - 1; ++i) {        // branch-free steady state
    int buf = i & 1;
    async_g2l_b128(gB + (i + 1) * 32, buf ? lB0 : lB1);  // tile i+1 in flight
    wait_async_le1();                       // own tile-i chunk has landed
    __syncthreads();                        // all waves' tile-i chunks visible
    int k0 = i * 32;
    __builtin_prefetch(A + (size_t)(row0 + nrel) * K + k0 + 32, 0, 1);
    compute(bbuf[buf], k0);
    __syncthreads();                        // previous buffer free for overwrite
  }
  wait_async_0();                           // peeled last k-step
  __syncthreads();
  compute(bbuf[(nt - 1) & 1], (nt - 1) * 32);

  int half = lane >> 4;
  #pragma unroll
  for (int t = 0; t < 4; ++t) {
    int col = col0 + t * 16 + nrel;
    #pragma unroll
    for (int r = 0; r < 8; ++r) {
      int row = row0 + r + 8 * half;        // C-matrix VGPR layout
      float v = acc[t][r];
      if (mode == 0) {
        outb[(size_t)row * Nn + col] = (__bf16)v;
      } else if (mode == 1) {
        outb[(size_t)col * M + row] = (__bf16)v;
      } else if (mode == 2) {
        if (row < Mvalid)
          outf[(size_t)row * Nn + col] =
              resid[(size_t)row * Nn + col] + v + bias[col];
      } else if (mode == 3) {
        outb[(size_t)row * Nn + col] = (__bf16)gelu_approx(v + bias[col]);
      } else {
        if (row < Mvalid)
          outf[(size_t)row * Nn + col] += v + bias[col];
      }
    }
  }
}

// ---------------- tri-block flash attention ----------------
// grid (NBLK, H, 2); 8 waves x 16 query rows = 128 queries per workgroup.
// Keys = blocks blk-1..blk+1 (768), masked by 0 <= global_key < N.
__global__ __launch_bounds__(256) void attn_kernel(
    const __bf16* __restrict__ Q, const __bf16* __restrict__ Kb,
    const __bf16* __restrict__ Vt, __bf16* __restrict__ O, int Nvalid) {
  __shared__ __align__(16) __bf16 pbuf[8][16 * 32];
  int lane = threadIdx.x & 31, w = threadIdx.x >> 5;
  int blk = blockIdx.x, hh = blockIdx.y, qhalf = blockIdx.z;
  int qrow0 = blk * BSZ + qhalf * 128 + w * 16;
  int half = lane >> 4, nrel = lane & 15;

  // Q fragments for the 4 head-dim chunks (hoisted out of key loop)
  bf16x16 qf[4];
  #pragma unroll
  for (int d = 0; d < 4; ++d)
    qf[d] = load_a_frag(Q, DD, qrow0, hh * HDQ + d * 32, lane);

  f32x8 oacc[8] = {};
  float mrow[8], lrow[8];
  #pragma unroll
  for (int r = 0; r < 8; ++r) { mrow[r] = -1e30f; lrow[r] = 0.f; }

  int kbase = (blk - 1) * BSZ;   // may be negative (masked away)

  for (int kc = 0; kc < 3 * BSZ; kc += 32) {
    // ---- S = Q K^T for 2 column tiles of 16 keys ----
    f32x8 sacc[2] = {};
    #pragma unroll
    for (int ct = 0; ct < 2; ++ct) {
      int keyLane = kbase + kc + ct * 16 + nrel;      // this lane's key column
      int keySafe = imin(imax(keyLane, 0), NP - 1);
      const __bf16* kp = Kb + (size_t)keySafe * DD + hh * HDQ + half * 8;
      Frag16 f[4];
      #pragma unroll
      for (int d = 0; d < 4; ++d) {         // clause of 8 global_load_b128
        f[d].q[0] = *(const u32x4*)(kp + d * 32);
        f[d].q[1] = *(const u32x4*)(kp + d * 32 + 16);
      }
      #pragma unroll
      for (int d = 0; d < 4; ++d) sacc[ct] = wmma_bf16(qf[d], f[d].v, sacc[ct]);
    }
    // ---- scale + mask (key validity is column-only: 0 <= g < N) ----
    #pragma unroll
    for (int ct = 0; ct < 2; ++ct) {
      int g = kbase + kc + ct * 16 + nrel;
      bool valid = (g >= 0) && (g < Nvalid);
      #pragma unroll
      for (int r = 0; r < 8; ++r)
        sacc[ct][r] = valid ? sacc[ct][r] * ATT_SCALE : -1e30f;
    }
    // ---- online softmax: row max across 32 keys (16 lanes/half + 2 tiles) ----
    #pragma unroll
    for (int r = 0; r < 8; ++r) {
      float v = fmaxf(sacc[0][r], sacc[1][r]);
      #pragma unroll
      for (int msk = 1; msk < 16; msk <<= 1)
        v = fmaxf(v, __shfl_xor(v, msk, 32));
      float mnew = fmaxf(mrow[r], v);
      float fct = __expf(mrow[r] - mnew);
      lrow[r] *= fct;
      #pragma unroll
      for (int t = 0; t < 8; ++t) oacc[t][r] *= fct;
      mrow[r] = mnew;
    }
    // ---- P = exp(S - m): accumulate partial denom, bounce through LDS ----
    __bf16* pb = pbuf[w];
    #pragma unroll
    for (int ct = 0; ct < 2; ++ct) {
      #pragma unroll
      for (int r = 0; r < 8; ++r) {
        float p = __expf(sacc[ct][r] - mrow[r]);
        lrow[r] += p;
        pb[(r + 8 * half) * 32 + ct * 16 + nrel] = (__bf16)p;  // row-major 16x32
      }
    }
    __syncthreads();
    // ---- O += P V : reload P in A layout, V^T gives contiguous B frags ----
    bf16x16 pf = load_a_frag(pb, 32, 0, 0, lane);
    int key0 = kbase + kc + half * 8;
    int keySafe0 = imin(imax(key0, 0), NP - 32);
    #pragma unroll
    for (int t = 0; t < 8; ++t) {
      int dcol = hh * HDQ + t * 16 + nrel;
      const __bf16* p = Vt + (size_t)dcol * NP + keySafe0;
      Frag16 f;
      f.q[0] = *(const u32x4*)(p);
      f.q[1] = *(const u32x4*)(p + 16);
      oacc[t] = wmma_bf16(pf, f.v, oacc[t]);
    }
    __syncthreads();
  }

  // ---- finalize: total denominator per row, store O/denom as bf16 ----
  #pragma unroll
  for (int r = 0; r < 8; ++r) {
    float v = lrow[r];
    #pragma unroll
    for (int msk = 1; msk < 16; msk <<= 1) v += __shfl_xor(v, msk, 32);
    lrow[r] = 1.0f / v;
  }
  #pragma unroll
  for (int t = 0; t < 8; ++t) {
    int col = hh * HDQ + t * 16 + nrel;
    #pragma unroll
    for (int r = 0; r < 8; ++r) {
      int row = qrow0 + r + 8 * half;
      O[(size_t)row * DD + col] = (__bf16)(oacc[t][r] * lrow[r]);
    }
  }
}

// ---------------- host orchestration ----------------
extern "C" void kernel_launch(void* const* d_in, const int* in_sizes, int n_in,
                              void* d_out, int out_size, void* d_ws, size_t ws_size,
                              hipStream_t stream) {
  const float* x      = (const float*)d_in[0];
  const float* gnc    = (const float*)d_in[1];
  // d_in[2] = mask (unused; mask is computed analytically)
  const float* wq     = (const float*)d_in[3];
  const float* wk     = (const float*)d_in[4];
  const float* wv     = (const float*)d_in[5];
  const float* wfin   = (const float*)d_in[6];
  const float* bfin   = (const float*)d_in[7];
  const float* wup    = (const float*)d_in[8];
  const float* bup    = (const float*)d_in[9];
  const float* wdn    = (const float*)d_in[10];
  const float* bdn    = (const float*)d_in[11];
  const float* wcond  = (const float*)d_in[12];
  const float* bcond  = (const float*)d_in[13];
  float* out = (float*)d_out;

  // workspace carve-up
  char* p = (char*)d_ws;
  auto take = [&](size_t bytes) { char* r = p; p += (bytes + 255) & ~(size_t)255; return r; };
  float*  so   = (float*)take(2 * DD * sizeof(float));
  __bf16* wqT  = (__bf16*)take((size_t)DD * DD * 2);
  __bf16* wkT  = (__bf16*)take((size_t)DD * DD * 2);
  __bf16* wvT  = (__bf16*)take((size_t)DD * DD * 2);
  __bf16* wfT  = (__bf16*)take((size_t)DD * DD * 2);
  __bf16* wupT = (__bf16*)take((size_t)FFW * DD * 2);
  __bf16* wdnT = (__bf16*)take((size_t)DD * FFW * 2);
  __bf16* h1   = (__bf16*)take((size_t)NP * DD * 2);
  __bf16* qb   = (__bf16*)take((size_t)NP * DD * 2);
  __bf16* kb   = (__bf16*)take((size_t)NP * DD * 2);
  __bf16* vT   = (__bf16*)take((size_t)NP * DD * 2);
  __bf16* aO   = (__bf16*)take((size_t)NP * DD * 2);
  __bf16* h2   = (__bf16*)take((size_t)NP * DD * 2);
  __bf16* uAct = (__bf16*)take((size_t)NP * FFW * 2);
  (void)ws_size; (void)in_sizes; (void)n_in; (void)out_size;

  // 1) conditioning vector
  cond_kernel<<<dim3(4), dim3(256), 0, stream>>>(gnc, wcond, bcond, so);

  // 2) weight casts/transposes to bf16 [out][in]
  transpose_cast<<<dim3((DD * DD) / 256), dim3(256), 0, stream>>>(wq,   wqT,  DD, DD);
  transpose_cast<<<dim3((DD * DD) / 256), dim3(256), 0, stream>>>(wk,   wkT,  DD, DD);
  transpose_cast<<<dim3((DD * DD) / 256), dim3(256), 0, stream>>>(wv,   wvT,  DD, DD);
  transpose_cast<<<dim3((DD * DD) / 256), dim3(256), 0, stream>>>(wfin, wfT,  DD, DD);
  transpose_cast<<<dim3((DD * FFW) / 256), dim3(256), 0, stream>>>(wup, wupT, DD, FFW);
  transpose_cast<<<dim3((FFW * DD) / 256), dim3(256), 0, stream>>>(wdn, wdnT, FFW, DD);

  // 3) h1 = cond(LN(x)), zero-padded to NP rows
  ln_cond<<<dim3(NP / 8), dim3(256), 0, stream>>>(x, so, h1, NN, NP);

  // 4) Q,K,V projections (V stored transposed for the P·V stage)
  dim3 g512((NP + 127) / 128, DD / 64);
  gemm_bf16<<<g512, dim3(256), 0, stream>>>(h1, wqT, qb, nullptr, nullptr, nullptr,
                                            NP, DD, DD, NP, 0);
  gemm_bf16<<<g512, dim3(256), 0, stream>>>(h1, wkT, kb, nullptr, nullptr, nullptr,
                                            NP, DD, DD, NP, 0);
  gemm_bf16<<<g512, dim3(256), 0, stream>>>(h1, wvT, vT, nullptr, nullptr, nullptr,
                                            NP, DD, DD, NP, 1);

  // 5) tri-block attention
  attn_kernel<<<dim3(NBLK, HH, 2), dim3(256), 0, stream>>>(qb, kb, vT, aO, NN);

  // 6) x1 = x + attnO @ w_final + b_final   (rows < N)
  gemm_bf16<<<g512, dim3(256), 0, stream>>>(aO, wfT, nullptr, out, x, bfin,
                                            NP, DD, DD, NN, 2);

  // 7) h2 = cond(LN(x1))
  ln_cond<<<dim3(NP / 8), dim3(256), 0, stream>>>(out, so, h2, NN, NP);

  // 8) u = gelu(h2 @ w_up + b_up)
  dim3 gUp((NP + 127) / 128, FFW / 64);
  gemm_bf16<<<gUp, dim3(256), 0, stream>>>(h2, wupT, uAct, nullptr, nullptr, bup,
                                           NP, FFW, DD, NP, 3);

  // 9) x_out = x1 + u @ w_down + b_down  (in-place accumulate, rows < N)
  gemm_bf16<<<g512, dim3(256), 0, stream>>>(uAct, wdnT, nullptr, out, nullptr, bdn,
                                            NP, DD, FFW, NN, 4);
}
